// GRVQ_2559800508669
// MI455X (gfx1250) — matrix operand order repaired and checked by reading, not compiled, not run on hardware
//
#include <hip/hip_runtime.h>

// Problem constants (from reference)
#define G   2
#define Q   8
#define KC  1024        // codes per codebook
#define CD  256         // codebook dim
#define DIM 1280
#define DG  (DIM / G)   // 640
#define BN  (8 * 4096)  // 32768 tokens
#define MT  (BN / 16)   // 2048 16-row tiles per group
#define WPB 8           // waves per block in k_rvq (8 x 16-token tiles)

typedef __bf16 bf16;
typedef __attribute__((ext_vector_type(16))) __bf16 v16bf;
typedef __attribute__((ext_vector_type(8)))  __bf16 v8bf;
typedef __attribute__((ext_vector_type(8)))  float  v8f;
typedef __attribute__((ext_vector_type(4)))  float  v4f;
typedef __attribute__((ext_vector_type(4)))  unsigned v4u;
typedef __attribute__((ext_vector_type(4)))  int    v4i;

#if defined(__has_builtin)
#if __has_builtin(__builtin_amdgcn_global_load_async_to_lds_b128)
#define HAVE_ASYNC_LDS 1
#endif
#endif

// 16-byte global -> LDS copy. Async (ASYNCcnt-tracked, no VGPR round trip) when
// the gfx1250 builtin exists; synchronous fallback otherwise.
// Builtin signature (per hipcc diagnostic): arg0 = int4 in AS(1) (global src),
// arg1 = int4 in AS(3) (LDS dst), then imm offset, imm cpol.
static __device__ __forceinline__ void async_cp16(void* l, const void* gsrc) {
#ifdef HAVE_ASYNC_LDS
  typedef __attribute__((address_space(1))) v4i* gp_t;
  typedef __attribute__((address_space(3))) v4i* lp_t;
  __builtin_amdgcn_global_load_async_to_lds_b128((gp_t)gsrc, (lp_t)l, 0, 0);
#else
  *(v4u*)l = *(const v4u*)gsrc;
#endif
}

static __device__ __forceinline__ void wait_async() {
#ifdef HAVE_ASYNC_LDS
#if __has_builtin(__builtin_amdgcn_s_wait_asynccnt)
  __builtin_amdgcn_s_wait_asynccnt(0);
#else
  asm volatile("s_wait_asynccnt 0" ::: "memory");
#endif
#endif
}

static __device__ __forceinline__ void wave_lds_fence() {
  asm volatile("s_wait_dscnt 0" ::: "memory");
}

// A-fragment (16xK=32, 16-bit): lane L -> row = L%16; lanes<16 hold K 0..15,
// lanes>=16 hold K 16..31. Two contiguous 16B loads per lane.
static __device__ __forceinline__ v16bf afrag_bf16(const bf16* row, int lane) {
  const int off = (lane < 16) ? 0 : 16;
  v8bf lo = *(const v8bf*)(row + off);
  v8bf hi = *(const v8bf*)(row + 8 + off);
  v16bf a;
#pragma unroll
  for (int i = 0; i < 8; ++i) { a[i] = lo[i]; a[8 + i] = hi[i]; }
  return a;
}

// B-fragment (K=32 x 16): lane L -> column L%16; lanes<16 hold K 0..15,
// lanes>=16 hold K 16..31: 16 *contiguous* source elements per lane.
static __device__ __forceinline__ v16bf pack16f(const float* p) {
  v4f x0 = *(const v4f*)(p);
  v4f x1 = *(const v4f*)(p + 4);
  v4f x2 = *(const v4f*)(p + 8);
  v4f x3 = *(const v4f*)(p + 12);
  v16bf b;
#pragma unroll
  for (int i = 0; i < 4; ++i) {
    b[i]      = (__bf16)x0[i];
    b[4 + i]  = (__bf16)x1[i];
    b[8 + i]  = (__bf16)x2[i];
    b[12 + i] = (__bf16)x3[i];
  }
  return b;
}

// ---------------- K0a: codebook prep: bf16 copy + (-0.5 * ||c||^2) -------------
__global__ void k_prep_cb(const float* __restrict__ cb, bf16* __restrict__ cbbf,
                          float* __restrict__ nh) {
  const int w = threadIdx.x >> 5, lane = threadIdx.x & 31;
  const int row = blockIdx.x * 4 + w;             // < G*Q*KC
  const float* src = cb + (size_t)row * CD + lane * 8;
  v4f a = *(const v4f*)src, b = *(const v4f*)(src + 4);
  float s = 0.f;
  v8bf ob;
#pragma unroll
  for (int i = 0; i < 4; ++i) {
    s += a[i] * a[i] + b[i] * b[i];
    ob[i]     = (__bf16)a[i];
    ob[4 + i] = (__bf16)b[i];
  }
  *(v8bf*)(cbbf + (size_t)row * CD + lane * 8) = ob;
#pragma unroll
  for (int o = 16; o > 0; o >>= 1) s += __shfl_xor(s, o, 32);
  if (lane == 0) nh[row] = -0.5f * s;
}

// ---------------- K0b: weight bf16 conversion ----------------------------------
__global__ void k_prep_w(const float* __restrict__ Win, const float* __restrict__ Wout,
                         bf16* __restrict__ winbf, bf16* __restrict__ woutbf) {
  const int i = blockIdx.x * blockDim.x + threadIdx.x;   // < G*CD*DG
  winbf[i]  = (__bf16)Win[i];
  woutbf[i] = (__bf16)Wout[i];
}

// ---------------- K1: input projection h[g,m,c] = x[m,g*DG+:] . Win[g,c,:] + bin
// WMMA with A = Win rows (contiguous), B = x^T (contiguous per lane), D = h^T.
__global__ void k_inproj(const float* __restrict__ x, const bf16* __restrict__ winbf,
                         const float* __restrict__ bin, float* __restrict__ h) {
  const int w = threadIdx.x >> 5, lane = threadIdx.x & 31;
  const int tile = blockIdx.x * 4 + w;
  const int g = tile / MT, mt = tile % MT;
  const int m0 = mt * 16, m = lane & 15;
  const int boff = (lane < 16) ? 0 : 16;
  const int hoff = (lane < 16) ? 0 : 8;
  const float* xrow = x + (size_t)(m0 + m) * DIM + g * DG;
  const v8f vzero = {0.f, 0.f, 0.f, 0.f, 0.f, 0.f, 0.f, 0.f};

  for (int half = 0; half < 2; ++half) {        // CD split in two to bound VGPRs
    v8f acc[8];
#pragma unroll
    for (int ct = 0; ct < 8; ++ct) acc[ct] = vzero;
    for (int ch = 0; ch < DG / 32; ++ch) {      // 20 K-chunks of 32
      const int d0 = ch * 32;
      v16bf bfr = pack16f(xrow + d0 + boff);    // fp32 -> bf16 on the fly
#pragma unroll
      for (int ct = 0; ct < 8; ++ct) {
        const int crow = half * 128 + ct * 16 + m;
        v16bf afr = afrag_bf16(winbf + ((size_t)(g * CD + crow)) * DG + d0, lane);
        acc[ct] = __builtin_amdgcn_wmma_f32_16x16x32_bf16(
            false, afr, false, bfr, (short)0, acc[ct], false, false);
      }
    }
#pragma unroll
    for (int ct = 0; ct < 8; ++ct) {
      const int c0 = half * 128 + ct * 16 + hoff;
      v4f bb0 = *(const v4f*)(bin + g * CD + c0);
      v4f bb1 = *(const v4f*)(bin + g * CD + c0 + 4);
      v4f o0 = {acc[ct][0] + bb0[0], acc[ct][1] + bb0[1],
                acc[ct][2] + bb0[2], acc[ct][3] + bb0[3]};
      v4f o1 = {acc[ct][4] + bb1[0], acc[ct][5] + bb1[1],
                acc[ct][6] + bb1[2], acc[ct][7] + bb1[3]};
      float* hrow = h + ((size_t)g * BN + m0 + m) * CD + c0;
      *(v4f*)hrow = o0;
      *(v4f*)(hrow + 4) = o1;
    }
  }
}

// ---------------- K2: fused ResidualVQ (all Q steps, residual lives in LDS) ----
// 8 waves/block share each 16-code codebook tile, double-buffered in LDS via
// async global->LDS copies; residual rows (fp32) stay in LDS for all Q steps.
// scores^T[code][m] via WMMA (A = codebook tile in LDS, B = res^T from LDS),
// accumulator seeded with -0.5*||c||^2  =>  argmax == argmin of euclidean dist.
__global__ void __launch_bounds__(32 * WPB)
k_rvq(const float* __restrict__ h, const bf16* __restrict__ cbbf,
      const float* __restrict__ nh, const float* __restrict__ cbf32,
      float* __restrict__ res, float* __restrict__ outIdx,
      float* __restrict__ ploss) {
  __shared__ __align__(16) float Lres[WPB][16][CD];     // 128 KB residual rows
  __shared__ __align__(16) bf16  Lcb[2][16][CD + 8];    // ~16.9 KB staged tiles
                                                        // (+8 pad -> 4-bank skew)
  const int tid = threadIdx.x;
  const int w = tid >> 5, lane = tid & 31;
  const int tile = blockIdx.x * WPB + w;                // MT % WPB == 0
  const int g = tile / MT, mt = tile % MT;
  const int m0 = mt * 16, m = lane & 15;
  const int hoff = (lane < 16) ? 0 : 8;
  const int boff = (lane < 16) ? 0 : 16;
  float (*L)[CD] = Lres[w];

  // Stage-copy role for codebook tiles: 256 threads x 2 x 16B = 8 KB tile.
  const int srow = tid >> 4;          // 0..15 : code row within tile
  const int scol = (tid & 15) * 16;   // element offset within row

  // Async-fill the 16 residual rows (h) into LDS.
  for (int r = 0; r < 16; ++r) {
    const float* src = h + ((size_t)g * BN + m0 + r) * CD + lane * 8;
    async_cp16(&L[r][lane * 8], src);
    async_cp16(&L[r][lane * 8 + 4], src + 4);
  }
  wait_async();
  wave_lds_fence();

  for (int q = 0; q < Q; ++q) {
    // Rebuild residual^T B-fragments from LDS (bf16 cvt).
    v16bf bfr[8];
#pragma unroll
    for (int ch = 0; ch < 8; ++ch)
      bfr[ch] = pack16f(&L[m][ch * 32 + boff]);

    const bf16*  cbq = cbbf + ((size_t)(g * Q + q)) * KC * CD;
    const float* nhq = nh   + ((size_t)(g * Q + q)) * KC;

    // Prologue: stage code tile 0 into buffer 0.
    {
      const bf16* gsrc = cbq + (size_t)srow * CD + scol;
      async_cp16(&Lcb[0][srow][scol], gsrc);
      async_cp16(&Lcb[0][srow][scol + 8], gsrc + 8);
    }
    wait_async();
    __syncthreads();

    float best = -3.4e38f;
    int   bidx = 0;
    for (int kt = 0; kt < KC / 16; ++kt) {
      const int cur = kt & 1;
      if (kt < KC / 16 - 1) {   // issue async stage of next tile, overlap WMMA
        const bf16* gsrc = cbq + (size_t)((kt + 1) * 16 + srow) * CD + scol;
        async_cp16(&Lcb[cur ^ 1][srow][scol], gsrc);
        async_cp16(&Lcb[cur ^ 1][srow][scol + 8], gsrc + 8);
      }
      const float* c2 = nhq + kt * 16 + hoff;
      v4f c2a = *(const v4f*)c2, c2b = *(const v4f*)(c2 + 4);
      v8f acc = {c2a[0], c2a[1], c2a[2], c2a[3], c2b[0], c2b[1], c2b[2], c2b[3]};
      const bf16* arow = &Lcb[cur][m][0];
#pragma unroll
      for (int ch = 0; ch < 8; ++ch) {
        v16bf afr = afrag_bf16(arow + ch * 32, lane);
        acc = __builtin_amdgcn_wmma_f32_16x16x32_bf16(
            false, afr, false, bfr[ch], (short)0, acc, false, false);
      }
#pragma unroll
      for (int r = 0; r < 8; ++r) {
        const float sv = acc[r];
        const int code = kt * 16 + hoff + r;
        if (sv > best) { best = sv; bidx = code; }
      }
      wait_async();       // own staged chunks landed in LDS
      __syncthreads();    // everyone done reading `cur`, everyone's stage visible
    }
    // Lanes L and L^16 hold the two code halves of residual row L%16: combine.
    {
      float ob = __shfl_xor(best, 16, 32);
      int   oi = __shfl_xor(bidx, 16, 32);
      if (ob > best || (ob == best && oi < bidx)) { best = ob; bidx = oi; }
    }
    if (lane < 16)
      outIdx[((size_t)g * BN + m0 + lane) * Q + q] = (float)bidx;

    // Residual update (fp32, in LDS, wave-local) + commitment loss partial.
    float lsum = 0.f;
    const float* cbq32 = cbf32 + ((size_t)(g * Q + q)) * KC * CD;
    for (int r = 0; r < 16; ++r) {
      const int code = __shfl(bidx, r, 32);
      const float* crow = cbq32 + (size_t)code * CD + lane * 8;
      v4f c0 = *(const v4f*)crow, c1 = *(const v4f*)(crow + 4);
      v4f r0 = *(v4f*)&L[r][lane * 8];
      v4f r1 = *(v4f*)&L[r][lane * 8 + 4];
#pragma unroll
      for (int i = 0; i < 4; ++i) {
        r0[i] -= c0[i]; r1[i] -= c1[i];
        lsum += r0[i] * r0[i] + r1[i] * r1[i];
      }
      *(v4f*)&L[r][lane * 8]     = r0;
      *(v4f*)&L[r][lane * 8 + 4] = r1;
    }
    wave_lds_fence();
#pragma unroll
    for (int o = 16; o > 0; o >>= 1) lsum += __shfl_xor(lsum, o, 32);
    if (lane == 0) ploss[(size_t)tile * Q + q] = lsum;   // deterministic partials
  }

  // Final residual back to global (qout = h - res for the output projection).
  for (int r = 0; r < 16; ++r) {
    float* dst = res + ((size_t)g * BN + m0 + r) * CD + lane * 8;
    *(v4f*)dst       = *(v4f*)&L[r][lane * 8];
    *(v4f*)(dst + 4) = *(v4f*)&L[r][lane * 8 + 4];
  }
}

// ---------------- K3: output projection out[m, g*DG+d] = (h-res).Wout^T + bout -
__global__ void k_outproj(const float* __restrict__ h, const float* __restrict__ res,
                          const bf16* __restrict__ woutbf, const float* __restrict__ bout,
                          float* __restrict__ out) {
  const int w = threadIdx.x >> 5, lane = threadIdx.x & 31;
  const int tile = blockIdx.x * 4 + w;
  const int g = tile / MT, mt = tile % MT;
  const int m0 = mt * 16, m = lane & 15;
  const int boff = (lane < 16) ? 0 : 16;
  const int hoff = (lane < 16) ? 0 : 8;

  // qout^T B-fragments (loaded once, reused for all 40 d-tiles).
  const float* hrow = h   + ((size_t)g * BN + m0 + m) * CD;
  const float* rrow = res + ((size_t)g * BN + m0 + m) * CD;
  v16bf bfr[8];
#pragma unroll
  for (int ch = 0; ch < 8; ++ch) {
    const int c0 = ch * 32 + boff;
    v4f a0 = *(const v4f*)(hrow + c0)      - *(const v4f*)(rrow + c0);
    v4f a1 = *(const v4f*)(hrow + c0 + 4)  - *(const v4f*)(rrow + c0 + 4);
    v4f a2 = *(const v4f*)(hrow + c0 + 8)  - *(const v4f*)(rrow + c0 + 8);
    v4f a3 = *(const v4f*)(hrow + c0 + 12) - *(const v4f*)(rrow + c0 + 12);
    v16bf b;
#pragma unroll
    for (int i = 0; i < 4; ++i) {
      b[i]      = (__bf16)a0[i];
      b[4 + i]  = (__bf16)a1[i];
      b[8 + i]  = (__bf16)a2[i];
      b[12 + i] = (__bf16)a3[i];
    }
    bfr[ch] = b;
  }

  for (int dt = 0; dt < DG / 16; ++dt) {
    const int dbase = dt * 16 + hoff;
    v4f bb0 = *(const v4f*)(bout + g * DG + dbase);
    v4f bb1 = *(const v4f*)(bout + g * DG + dbase + 4);
    v8f acc = {bb0[0], bb0[1], bb0[2], bb0[3], bb1[0], bb1[1], bb1[2], bb1[3]};
#pragma unroll
    for (int ch = 0; ch < 8; ++ch) {
      v16bf afr = afrag_bf16(
          woutbf + ((size_t)g * DG + dt * 16 + m) * CD + ch * 32, lane);
      acc = __builtin_amdgcn_wmma_f32_16x16x32_bf16(
          false, afr, false, bfr[ch], (short)0, acc, false, false);
    }
    float* orow = out + (size_t)(m0 + m) * DIM + g * DG + dbase;
    v4f o0 = {acc[0], acc[1], acc[2], acc[3]};
    v4f o1 = {acc[4], acc[5], acc[6], acc[7]};
    *(v4f*)orow = o0;
    *(v4f*)(orow + 4) = o1;
  }
}

// ---------------- K4: deterministic loss reduction -----------------------------
__global__ void k_loss(const float* __restrict__ ploss, float* __restrict__ outL) {
  __shared__ float red[256];
  const int gq = blockIdx.x;           // g*Q + q (matches reference (g,Q) layout)
  const int g = gq / Q, q = gq % Q;
  float s = 0.f;
  for (int t = threadIdx.x; t < MT; t += 256)
    s += ploss[((size_t)(g * MT + t)) * Q + q];
  red[threadIdx.x] = s;
  __syncthreads();
  for (int o = 128; o > 0; o >>= 1) {
    if (threadIdx.x < o) red[threadIdx.x] += red[threadIdx.x + o];
    __syncthreads();
  }
  if (threadIdx.x == 0)
    outL[gq] = red[0] * (1.0f / ((float)BN * (float)CD));
}

// -------------------------------------------------------------------------------
extern "C" void kernel_launch(void* const* d_in, const int* in_sizes, int n_in,
                              void* d_out, int out_size, void* d_ws, size_t ws_size,
                              hipStream_t stream) {
  const float* x    = (const float*)d_in[0];
  const float* Win  = (const float*)d_in[1];
  const float* bin  = (const float*)d_in[2];
  const float* Wout = (const float*)d_in[3];
  const float* bout = (const float*)d_in[4];
  const float* cb   = (const float*)d_in[5];
  float* out = (float*)d_out;

  // Workspace layout (~137.5 MiB total, all 16B-aligned offsets):
  char* ws = (char*)d_ws;
  size_t off = 0;
  float* h      = (float*)(ws + off); off += (size_t)G * BN * CD * 4;     // 64 MiB
  float* res    = (float*)(ws + off); off += (size_t)G * BN * CD * 4;     // 64 MiB
  bf16*  cbbf   = (bf16*)(ws + off);  off += (size_t)G * Q * KC * CD * 2; // 8 MiB
  float* nh     = (float*)(ws + off); off += (size_t)G * Q * KC * 4;
  bf16*  winbf  = (bf16*)(ws + off);  off += (size_t)G * CD * DG * 2;
  bf16*  woutbf = (bf16*)(ws + off);  off += (size_t)G * DG * CD * 2;
  float* ploss  = (float*)(ws + off); off += (size_t)G * MT * Q * 4;

  float* outQ   = out;                               // (B,N,DIM)
  float* outIdx = out + (size_t)BN * DIM;            // (G,B,N,Q) as float
  float* outL   = outIdx + (size_t)G * BN * Q;       // (G,Q)

  k_prep_cb<<<G * Q * KC / 4, 128, 0, stream>>>(cb, cbbf, nh);
  k_prep_w<<<(G * CD * DG) / 256, 256, 0, stream>>>(Win, Wout, winbf, woutbf);
  k_inproj<<<(G * MT) / 4, 128, 0, stream>>>(x, winbf, bin, h);
  k_rvq<<<(G * MT) / WPB, 32 * WPB, 0, stream>>>(h, cbbf, nh, cb, res, outIdx, ploss);
  k_outproj<<<(G * MT) / 4, 128, 0, stream>>>(h, res, woutbf, bout, outQ);
  k_loss<<<G * Q, 256, 0, stream>>>(ploss, outL);
}
